// NoGatingAttention_18176301597139
// MI455X (gfx1250) — matrix operand order repaired and checked by reading, not compiled, not run on hardware
//
#include <hip/hip_runtime.h>

typedef __bf16 bf16t;
typedef __attribute__((ext_vector_type(16))) __bf16 bf16x16;
typedef __attribute__((ext_vector_type(8))) float v8f;
typedef uint32_t u32x4 __attribute__((ext_vector_type(4)));
typedef uint32_t u32x8 __attribute__((ext_vector_type(8)));

namespace {
constexpr int kB = 128, kQ = 384, kK = 384, kAD = 256, kMD = 256;
constexpr int kH = 8, kKD = 32, kVD = 32, kOUT = 256;
}

__device__ __forceinline__ v8f wmma_bf16(bf16x16 a, bf16x16 b, v8f c) {
  // D = A(16x32 bf16) * B(32x16 bf16) + C(16x16 f32)
  return __builtin_amdgcn_wmma_f32_16x16x32_bf16(false, a, false, b, (short)0, c,
                                                 false, false);
}

// A-matrix fragment (16x32) from a row-major 32-element bf16 row.
// Lane L holds row m=L%16; VGPR r holds K pair
// K = r*2 + p + (L/16)*16 (contiguous, matches B fragments).
__device__ __forceinline__ bf16x16 a_frag(const bf16t* row, int khalf) {
  bf16x16 a;
#pragma unroll
  for (int r = 0; r < 8; ++r) {
    const int k = (r << 1) + (khalf << 1);
    a[2 * r] = row[k];
    a[2 * r + 1] = row[k + 1];
  }
  return a;
}

// Same fragment but sourced from an f32 row (TDM stages raw f32); the
// bf16 conversion rides along as v_cvt_pk_bf16_f32.
__device__ __forceinline__ bf16x16 a_frag_f32(const float* row, int khalf) {
  bf16x16 a;
#pragma unroll
  for (int r = 0; r < 8; ++r) {
    const int k = (r << 1) + (khalf << 1);
    a[2 * r] = (bf16t)row[k];
    a[2 * r + 1] = (bf16t)row[k + 1];
  }
  return a;
}

__device__ __forceinline__ v8f zero8() {
  v8f z = {0.f, 0.f, 0.f, 0.f, 0.f, 0.f, 0.f, 0.f};
  return z;
}

// Tensor Data Mover: DMA one 384x64 f32 tile (row stride 256 f32) from
// global memory into LDS. D# per CDNA5 ISA.
__device__ __forceinline__ void tdm_load_tile_384x64_f32(uint32_t lds_addr,
                                                         const float* gsrc) {
  const uint64_t ga = (uint64_t)(uintptr_t)gsrc;
  u32x4 g0;
  g0[0] = 1u;                     // count=1 (valid user descriptor)
  g0[1] = lds_addr;               // D#.lds_addr (bytes)
  g0[2] = (uint32_t)ga;           // global_addr[31:0]
  g0[3] = (uint32_t)((ga >> 32) & 0x01ffffffu) | (2u << 30);  // addr[56:32]|type=2
  u32x8 g1;
  g1[0] = 2u << 16;               // workgroup_mask=0, data_size=2 (4 bytes)
  g1[1] = 64u << 16;              // tensor_dim0 = 64 (atomic barrier addr = 0)
  g1[2] = 384u << 16;             // tensor_dim0 hi=0 | tensor_dim1 = 384
  g1[3] = 64u << 16;              // tensor_dim1 hi=0 | tile_dim0 = 64
  g1[4] = 384u;                   // tile_dim1 = 384, tile_dim2 = 0
  g1[5] = 256u;                   // tensor_dim0_stride = 256 elements
  g1[6] = 0u;
  g1[7] = 0u;
  asm volatile("tensor_load_to_lds %0, %1" ::"s"(g0), "s"(g1) : "memory");
}

// One workgroup per (batch, head). 256 threads = 8 wave32.
__global__ __launch_bounds__(256) void fused_attn_kernel(
    const float* __restrict__ q_data, const float* __restrict__ m_data,
    const float* __restrict__ bias, const float* __restrict__ nbias,
    const float* __restrict__ query_w, const float* __restrict__ key_w,
    const float* __restrict__ value_w, bf16t* __restrict__ wavg) {
  extern __shared__ char smem[];
  bf16t* wkT = (bf16t*)smem;              // [32][256]  key_w^T   (bf16)
  bf16t* wvT = wkT + kKD * kMD;           // [32][256]  value_w^T
  bf16t* wqT = wvT + kKD * kMD;           // [32][256]  query_w^T (pre-scaled)
  float* stagef = (float*)(wqT + kKD * kAD);  // 2 x [384][64] f32 (TDM dest)
  bf16t* kbf = (bf16t*)(stagef + 2 * kK * 64);  // [384][32] K row-major
  bf16t* vT = kbf + kK * kKD;             // [32][384]  V^T
  bf16t* qbf = vT + kVD * kK;             // [384][32]  Q row-major
  bf16t* plds = (bf16t*)stagef;           // [8][16][32] overlay (phase 3 only)

  const int tid = threadIdx.x;
  const int wave = tid >> 5;
  const int lane = tid & 31;
  const int n = lane & 15;       // column within 16-wide tile
  const int hi = lane >> 4;      // half-wave select
  const int khalf = hi << 3;     // A-fragment K offset
  const int half16 = hi << 4;    // B-fragment K offset
  const int b = blockIdx.x >> 3;
  const int h = blockIdx.x & 7;

  const float* m_src = m_data + (size_t)b * kK * kMD;
  const float* q_src = q_data + (size_t)b * kQ * kAD;
  const uint32_t stage_lds0 = (uint32_t)(uintptr_t)stagef;
  const uint32_t stage_lds1 = (uint32_t)(uintptr_t)(stagef + kK * 64);

  // Kick off the first DMA before anything else so it overlaps weight staging.
  if (wave == 0) tdm_load_tile_384x64_f32(stage_lds0, m_src);

  // ---- per-head weights -> transposed bf16 in LDS (scale folded into Q) ----
  const float scale = 0.17677669529663687f;  // 32^-0.5
  for (int idx = tid; idx < kAD * kKD; idx += 256) {
    const int a = idx >> 5;
    const int c = idx & 31;
    const int g = (a * kH + h) * kKD + c;
    wkT[c * kMD + a] = (bf16t)key_w[g];
    wvT[c * kMD + a] = (bf16t)value_w[g];
    wqT[c * kAD + a] = (bf16t)(query_w[g] * scale);
  }

  // ---- K and V projections: m_data[b] (384x256) @ W (256x32) ----
  v8f accK[3][2], accV[3][2];
#pragma unroll
  for (int i = 0; i < 3; ++i)
#pragma unroll
    for (int j = 0; j < 2; ++j) {
      accK[i][j] = zero8();
      accV[i][j] = zero8();
    }

#pragma unroll 1
  for (int ac = 0; ac < 4; ++ac) {
    __syncthreads();  // everyone done with the buffer the next DMA will fill
    if (wave == 0) {
      if (ac < 3)
        tdm_load_tile_384x64_f32(((ac + 1) & 1) ? stage_lds1 : stage_lds0,
                                 m_src + (ac + 1) * 64);
      else  // pre-issue the first q_data chunk while we compute the last K/V one
        tdm_load_tile_384x64_f32(stage_lds0, q_src);
      __builtin_amdgcn_s_wait_tensorcnt(1);  // chunk `ac` has landed
    }
    __syncthreads();
    const float* st = stagef + (ac & 1) * kK * 64;
    __builtin_prefetch(q_src + (size_t)tid * 64, 0, 1);  // warm GL2 for q chunks
#pragma unroll
    for (int kk = 0; kk < 64; kk += 32) {
#pragma unroll
      for (int mtl = 0; mtl < 3; ++mtl) {
        const int mt = wave * 3 + mtl;
        const bf16x16 a = a_frag_f32(&st[(mt * 16 + n) * 64 + kk], khalf);
        const int abase = ac * 64 + kk;
#pragma unroll
        for (int nt = 0; nt < 2; ++nt) {
          const bf16x16 bk =
              *(const bf16x16*)&wkT[(nt * 16 + n) * kMD + abase + half16];
          accK[mtl][nt] = wmma_bf16(a, bk, accK[mtl][nt]);
          const bf16x16 bv =
              *(const bf16x16*)&wvT[(nt * 16 + n) * kMD + abase + half16];
          accV[mtl][nt] = wmma_bf16(a, bv, accV[mtl][nt]);
        }
      }
    }
  }
  // C layout: lane holds (m = r + 8*hi, n). K row-major, V transposed.
#pragma unroll
  for (int mtl = 0; mtl < 3; ++mtl)
#pragma unroll
    for (int nt = 0; nt < 2; ++nt)
#pragma unroll
      for (int r = 0; r < 8; ++r) {
        const int M = (wave * 3 + mtl) * 16 + r + (hi << 3);
        const int c = nt * 16 + n;
        kbf[M * kKD + c] = (bf16t)accK[mtl][nt][r];
        vT[c * kK + M] = (bf16t)accV[mtl][nt][r];
      }

  // ---- Q projection: q_data[b] (384x256) @ Wq (256x32) ----
  v8f accQ[3][2];
#pragma unroll
  for (int i = 0; i < 3; ++i)
#pragma unroll
    for (int j = 0; j < 2; ++j) accQ[i][j] = zero8();

#pragma unroll 1
  for (int ac = 0; ac < 4; ++ac) {
    __syncthreads();
    if (wave == 0) {
      if (ac < 3) {
        tdm_load_tile_384x64_f32(((ac + 1) & 1) ? stage_lds1 : stage_lds0,
                                 q_src + (ac + 1) * 64);
        __builtin_amdgcn_s_wait_tensorcnt(1);
      } else {
        __builtin_amdgcn_s_wait_tensorcnt(0);
      }
    }
    __syncthreads();
    const float* st = stagef + (ac & 1) * kK * 64;
#pragma unroll
    for (int kk = 0; kk < 64; kk += 32) {
#pragma unroll
      for (int mtl = 0; mtl < 3; ++mtl) {
        const int mt = wave * 3 + mtl;
        const bf16x16 a = a_frag_f32(&st[(mt * 16 + n) * 64 + kk], khalf);
        const int abase = ac * 64 + kk;
#pragma unroll
        for (int nt = 0; nt < 2; ++nt) {
          const bf16x16 bq =
              *(const bf16x16*)&wqT[(nt * 16 + n) * kAD + abase + half16];
          accQ[mtl][nt] = wmma_bf16(a, bq, accQ[mtl][nt]);
        }
      }
    }
  }
#pragma unroll
  for (int mtl = 0; mtl < 3; ++mtl)
#pragma unroll
    for (int nt = 0; nt < 2; ++nt)
#pragma unroll
      for (int r = 0; r < 8; ++r) {
        const int M = (wave * 3 + mtl) * 16 + r + (hi << 3);
        qbf[M * kKD + nt * 16 + n] = (bf16t)accQ[mtl][nt][r];
      }
  __syncthreads();  // stage buffers dead from here; plds overlay becomes live

  // ---- flash attention: each wave owns 3 q-tiles, streams 32-wide k-blocks
  const float* nb_h = nbias + (size_t)h * kQ * kK;
  const float* bias_b = bias + (size_t)b * kK;
  bf16t* pl = plds + wave * (16 * 32);

#pragma unroll 1
  for (int qi = 0; qi < 3; ++qi) {
    const int qt = wave + qi * 8;
    const int q0 = qt * 16;
    const bf16x16 aq = a_frag(&qbf[(q0 + n) * kKD], khalf);
    float mrun[8], lrun[8];
#pragma unroll
    for (int r = 0; r < 8; ++r) {
      mrun[r] = -1.0e30f;
      lrun[r] = 0.f;
    }
    v8f O0 = zero8(), O1 = zero8();

#pragma unroll 1
    for (int kb = 0; kb < 12; ++kb) {
      const int k0 = kb * 32;
      // S = Q @ K^T : B fragment column j is K row (k0+j), contiguous in c.
      const bf16x16 bk0 = *(const bf16x16*)&kbf[(k0 + n) * kKD + half16];
      const bf16x16 bk1 = *(const bf16x16*)&kbf[(k0 + 16 + n) * kKD + half16];
      v8f s0 = wmma_bf16(aq, bk0, zero8());
      v8f s1 = wmma_bf16(aq, bk1, zero8());

      const float bias0 = bias_b[k0 + n];
      const float bias1 = bias_b[k0 + 16 + n];
      const float* nbq = nb_h + (size_t)q0 * kK + k0;
#pragma unroll
      for (int r = 0; r < 8; ++r) {
        const int m = r + (hi << 3);
        s0[r] += bias0 + nbq[m * kK + n];
        s1[r] += bias1 + nbq[m * kK + 16 + n];
      }
      // online softmax: row stats live in the 16-lane group owning that row
#pragma unroll
      for (int r = 0; r < 8; ++r) {
        float t = fmaxf(s0[r], s1[r]);
        t = fmaxf(t, __shfl_xor(t, 1, 32));
        t = fmaxf(t, __shfl_xor(t, 2, 32));
        t = fmaxf(t, __shfl_xor(t, 4, 32));
        t = fmaxf(t, __shfl_xor(t, 8, 32));
        const float mnew = fmaxf(mrun[r], t);
        const float alpha = __expf(mrun[r] - mnew);
        mrun[r] = mnew;
        s0[r] = __expf(s0[r] - mnew);
        s1[r] = __expf(s1[r] - mnew);
        float rs = s0[r] + s1[r];
        rs += __shfl_xor(rs, 1, 32);
        rs += __shfl_xor(rs, 2, 32);
        rs += __shfl_xor(rs, 4, 32);
        rs += __shfl_xor(rs, 8, 32);
        lrun[r] = lrun[r] * alpha + rs;
        O0[r] *= alpha;
        O1[r] *= alpha;
      }
      // P (C layout, f32) -> wave-private LDS as bf16, reload as A fragment
#pragma unroll
      for (int r = 0; r < 8; ++r) {
        const int m = r + (hi << 3);
        pl[m * 32 + n] = (bf16t)s0[r];
        pl[m * 32 + 16 + n] = (bf16t)s1[r];
      }
      asm volatile("s_wait_dscnt 0" ::: "memory");
      const bf16x16 ap = a_frag(&pl[n * 32], khalf);
      // O += P(16x32) @ V(32x16): B fragment contiguous along k in V^T rows
      const bf16x16 bv0 = *(const bf16x16*)&vT[n * kK + k0 + half16];
      const bf16x16 bv1 = *(const bf16x16*)&vT[(16 + n) * kK + k0 + half16];
      O0 = wmma_bf16(ap, bv0, O0);
      O1 = wmma_bf16(ap, bv1, O1);
    }
    // normalize and emit weighted average as bf16 [B*Q, H*VD]
#pragma unroll
    for (int r = 0; r < 8; ++r) {
      const float inv = 1.f / lrun[r];
      const int m = r + (hi << 3);
      const size_t row = (size_t)b * kQ + q0 + m;
      wavg[row * (kH * kVD) + h * kVD + n] = (bf16t)(O0[r] * inv);
      wavg[row * (kH * kVD) + h * kVD + 16 + n] = (bf16t)(O1[r] * inv);
    }
  }
}

// Output projection: out[bq, o] = wavg[bq, :256] @ output_w[:256, o] + b[o]
__global__ __launch_bounds__(256) void out_proj_kernel(
    const bf16t* __restrict__ wavg, const float* __restrict__ output_w,
    const float* __restrict__ output_b, float* __restrict__ out) {
  extern __shared__ char smem2[];
  bf16t* woT = (bf16t*)smem2;  // [256][256] : woT[o][hc]
  const int tid = threadIdx.x;
  const int wave = tid >> 5;
  const int lane = tid & 31;
  const int n = lane & 15;
  const int hi = lane >> 4;
  const int khalf = hi << 3;
  const int half16 = hi << 4;

  for (int idx = tid; idx < 256 * 256; idx += 256) {
    const int hc = idx >> 8;
    const int o = idx & 255;
    woT[o * 256 + hc] = (bf16t)output_w[idx];
  }
  __syncthreads();

  const int m0 = blockIdx.x * 128 + wave * 16;
  const bf16t* arow = wavg + (size_t)(m0 + n) * 256;
  bf16x16 af[8];
#pragma unroll
  for (int kk = 0; kk < 8; ++kk) af[kk] = a_frag(arow + kk * 32, khalf);

#pragma unroll 1
  for (int nt = 0; nt < 16; ++nt) {
    v8f c = zero8();
#pragma unroll
    for (int kk = 0; kk < 8; ++kk) {
      const bf16x16 bw =
          *(const bf16x16*)&woT[(nt * 16 + n) * 256 + kk * 32 + half16];
      c = wmma_bf16(af[kk], bw, c);
    }
#pragma unroll
    for (int r = 0; r < 8; ++r) {
      const int o = nt * 16 + n;
      out[(size_t)(m0 + r + (hi << 3)) * kOUT + o] = c[r] + output_b[o];
    }
  }
}

extern "C" void kernel_launch(void* const* d_in, const int* in_sizes, int n_in,
                              void* d_out, int out_size, void* d_ws,
                              size_t ws_size, hipStream_t stream) {
  (void)in_sizes; (void)n_in; (void)out_size; (void)ws_size;
  const float* q_data = (const float*)d_in[0];
  const float* m_data = (const float*)d_in[1];
  const float* bias = (const float*)d_in[2];
  const float* nbias = (const float*)d_in[3];
  const float* query_w = (const float*)d_in[4];
  const float* key_w = (const float*)d_in[5];
  const float* value_w = (const float*)d_in[6];
  const float* output_w = (const float*)d_in[7];
  const float* output_b = (const float*)d_in[8];
  bf16t* wavg = (bf16t*)d_ws;  // [B*Q, H*VD] bf16 = 24 MB
  float* out = (float*)d_out;

  // LDS: 48 KB weights + 2x96 KB f32 TDM stage + 72 KB K/V^T/Q = 312 KB
  // (within the 320 KB/WGP budget; P-scratch overlays the stage region).
  const size_t smem1 =
      (size_t)(3 * 32 * 256 + 384 * 32 + 32 * 384 + 384 * 32) * sizeof(bf16t) +
      (size_t)2 * 384 * 64 * sizeof(float);
  fused_attn_kernel<<<kB * kH, 256, smem1, stream>>>(
      q_data, m_data, bias, nbias, query_w, key_w, value_w, wavg);

  const size_t smem2sz = (size_t)256 * 256 * sizeof(bf16t);
  out_proj_kernel<<<(kB * kQ) / 128, 256, smem2sz, stream>>>(wavg, output_w,
                                                             output_b, out);
}
